// GATNet_51462298140965
// MI455X (gfx1250) — compile-verified, matrix operand order checked by reading, not run on hardware
//
#include <hip/hip_runtime.h>
#include <math.h>

typedef __attribute__((ext_vector_type(16))) _Float16 v16h;
typedef __attribute__((ext_vector_type(8)))  _Float16 v8h;
typedef __attribute__((ext_vector_type(8)))  float    v8f;

#define NNODES 100000
#define NEDGES 1600000

union V16 { v16h v; v8h h[2]; };

// ---------------- casts ----------------
__global__ void cast_f32_f16(const float* __restrict__ x, _Float16* __restrict__ y, int n) {
  int i = blockIdx.x * blockDim.x + threadIdx.x;
  if (i < n) y[i] = (_Float16)x[i];
}

// W: [K, Nout] f32 row-major  ->  Wt: [Nout, K] f16 (so B columns are contiguous)
__global__ void transpose_cast_W(const float* __restrict__ W, _Float16* __restrict__ Wt,
                                 int K, int Nout) {
  int i = blockIdx.x * blockDim.x + threadIdx.x;
  if (i >= K * Nout) return;
  int n = i / K, k = i - n * K;
  Wt[n * K + k] = (_Float16)W[k * Nout + n];
}

// ---------------- WMMA GEMM: Z[N,Fout] = A[N,K] * W[K,Fout] ----------------
// K is compile-time so the a[] register array is statically indexed (no v_movrels).
// One wave per 16-row tile; wave-uniform guard keeps EXEC all-ones at each WMMA.
template <int K>
__global__ void gemm_wmma(const _Float16* __restrict__ A, const _Float16* __restrict__ Bt,
                          float* __restrict__ Z, int nRowTiles, int Fout) {
  int wave = blockIdx.x * (blockDim.x >> 5) + (threadIdx.x >> 5);
  if (wave >= nRowTiles) return;
  int lane = threadIdx.x & 31;
  int hi   = lane >> 4;                          // half-wave select
  int r    = lane & 15;
  constexpr int NK = K / 32;

  // A-matrix 16x32 f16 layout: lanes 0-15 row M=r hold K 0..7 / 16..23,
  // lanes 16-31 row M=r hold K 8..15 / 24..31.
  V16 a[NK];
  const _Float16* arow = A + (size_t)(wave * 16 + r) * K;
#pragma unroll
  for (int ks = 0; ks < NK; ++ks) {
    int k0 = ks * 32 + hi * 8;
    a[ks].h[0] = *(const v8h*)(arow + k0);
    a[ks].h[1] = *(const v8h*)(arow + k0 + 16);
  }

  int nCT = Fout >> 4;
  for (int ct = 0; ct < nCT; ++ct) {
    const _Float16* brow = Bt + (size_t)(ct * 16 + r) * K;  // column N=r of B
    v8f c = {};
#pragma unroll
    for (int ks = 0; ks < NK; ++ks) {
      int k0 = ks * 32 + hi * 8;
      V16 b;
      b.h[0] = *(const v8h*)(brow + k0);
      b.h[1] = *(const v8h*)(brow + k0 + 16);
      c = __builtin_amdgcn_wmma_f32_16x16x32_f16(false, a[ks].v, false, b.v,
                                                 (short)0, c, false, false);
    }
    // D layout: lane = n + (m>=8)*16 ; VGPR i -> M = i (+8 for upper half-wave)
    float* zt = Z + (size_t)(wave * 16) * Fout + ct * 16 + r;
#pragma unroll
    for (int i = 0; i < 8; ++i) zt[(size_t)(i + hi * 8) * Fout] = c[i];
  }
}

// ---------------- attention coefficients el/er ----------------
__global__ void attn_coeff(const float* __restrict__ Z, const float* __restrict__ al,
                           const float* __restrict__ ar, float* __restrict__ el,
                           float* __restrict__ er, int N, int H, int D) {
  int i = blockIdx.x * blockDim.x + threadIdx.x;
  if (i >= N * H) return;
  int n = i / H, h = i - n * H;
  const float* zr  = Z + (size_t)n * H * D + h * D;
  const float* alr = al + h * D;
  const float* arr = ar + h * D;
  float sl = 0.f, sr = 0.f;
  for (int d = 0; d < D; ++d) { float v = zr[d]; sl += v * alr[d]; sr += v * arr[d]; }
  el[i] = sl; er[i] = sr;
}

// ---------------- inits ----------------
__global__ void init_ms(float* __restrict__ m, float* __restrict__ s, int n) {
  int i = blockIdx.x * blockDim.x + threadIdx.x;
  if (i < n) { m[i] = -INFINITY; s[i] = 0.f; }
}
__global__ void init_agg_bias(float* __restrict__ agg, const float* __restrict__ b,
                              int total, int fmask) {
  int i = blockIdx.x * blockDim.x + threadIdx.x;
  if (i < total) agg[i] = b[i & fmask];
}
__global__ void zero_f32(float* __restrict__ p, int n) {
  int i = blockIdx.x * blockDim.x + threadIdx.x;
  if (i < n) p[i] = 0.f;
}

// ---------------- edge phase ----------------
__device__ __forceinline__ float lrelu(float v) { return v > 0.f ? v : 0.2f * v; }

__device__ __forceinline__ void atomicMaxF(float* addr, float v) {
  if (v >= 0.f) atomicMax((int*)addr, __float_as_int(v));
  else          atomicMin((unsigned int*)addr, __float_as_uint(v));
}

__global__ void edge_max(const int* __restrict__ src, const int* __restrict__ dst,
                         const float* __restrict__ el, const float* __restrict__ er,
                         float* __restrict__ m, int E, int H) {
  int i = blockIdx.x * blockDim.x + threadIdx.x;
  if (i >= E * H) return;
  int e = i / H, h = i - e * H;
  int sN = src[e], dN = dst[e];
  float v = lrelu(el[sN * H + h] + er[dN * H + h]);
  atomicMaxF(&m[dN * H + h], v);
}

__global__ void edge_expsum(const int* __restrict__ src, const int* __restrict__ dst,
                            const float* __restrict__ el, const float* __restrict__ er,
                            const float* __restrict__ m, float* __restrict__ exb,
                            float* __restrict__ s, int E, int H) {
  int i = blockIdx.x * blockDim.x + threadIdx.x;
  if (i >= E * H) return;
  int e = i / H, h = i - e * H;
  int sN = src[e], dN = dst[e];
  float v  = lrelu(el[sN * H + h] + er[dN * H + h]);
  float mm = m[dN * H + h];
  if (mm < -1e37f) mm = 0.f;                    // zero-in-degree guard (mirrors ref)
  float ex = __expf(v - mm);
  exb[i] = ex;
  atomicAdd(&s[dN * H + h], ex);
}

// one thread per (edge, feature): coalesced z[src] gather, coalesced agg[dst] atomics
__global__ void edge_agg(const int* __restrict__ src, const int* __restrict__ dst,
                         const float* __restrict__ exb, const float* __restrict__ s,
                         const float* __restrict__ Z, float* __restrict__ agg,
                         long long total, int fshift, int fmask, int dshift, int H) {
  long long i = (long long)blockIdx.x * blockDim.x + threadIdx.x;
  if (i >= total) return;
  int e = (int)(i >> fshift);
  int f = (int)i & fmask;
  int h = f >> dshift;
  int sN = src[e], dN = dst[e];
  float sv = s[dN * H + h];
  float w  = exb[(size_t)e * H + h] / (sv > 0.f ? sv : 1.f);
  atomicAdd(&agg[((size_t)dN << fshift) + f], w * Z[((size_t)sN << fshift) + f]);
}

// ---------------- batch norm ----------------
__global__ void bn_partial(const float* __restrict__ x, float* __restrict__ sum,
                           float* __restrict__ sq, int N, int C, int rowsPerBlock) {
  int c  = threadIdx.x;                         // blockDim.x == C
  int r0 = blockIdx.x * rowsPerBlock;
  int r1 = r0 + rowsPerBlock; if (r1 > N) r1 = N;
  float s = 0.f, q = 0.f;
  for (int r = r0; r < r1; ++r) { float v = x[(size_t)r * C + c]; s += v; q += v * v; }
  atomicAdd(&sum[c], s);
  atomicAdd(&sq[c], q);
}

__global__ void bn_stats(const float* __restrict__ sum, const float* __restrict__ sq,
                         const float* __restrict__ g, const float* __restrict__ be,
                         float* __restrict__ scale, float* __restrict__ shift, int N, int C) {
  int c = threadIdx.x;
  if (c >= C) return;
  float mu  = sum[c] / (float)N;
  float var = sq[c] / (float)N - mu * mu;
  float sc  = g[c] * rsqrtf(var + 1e-5f);
  scale[c] = sc;
  shift[c] = be[c] - mu * sc;
}

__global__ void bn_apply(float* __restrict__ x, const float* __restrict__ scale,
                         const float* __restrict__ shift, int total, int cmask) {
  int i = blockIdx.x * blockDim.x + threadIdx.x;
  if (i < total) { int c = i & cmask; x[i] = x[i] * scale[c] + shift[c]; }
}

// ---------------- log_softmax over 64 cols, one wave32 per node ----------------
__global__ void log_softmax64(const float* __restrict__ x, float* __restrict__ out, int N) {
  int node = blockIdx.x * (blockDim.x >> 5) + (threadIdx.x >> 5);
  if (node >= N) return;
  int lane = threadIdx.x & 31;
  const float* xr = x + (size_t)node * 64;
  float v0 = xr[lane], v1 = xr[lane + 32];
  float mx = fmaxf(v0, v1);
  for (int off = 16; off; off >>= 1) mx = fmaxf(mx, __shfl_xor(mx, off, 32));
  float se = __expf(v0 - mx) + __expf(v1 - mx);
  for (int off = 16; off; off >>= 1) se += __shfl_xor(se, off, 32);
  float l = mx + __logf(se);
  out[(size_t)node * 64 + lane]      = v0 - l;
  out[(size_t)node * 64 + lane + 32] = v1 - l;
}

// ---------------- host orchestration ----------------
static inline int gblk(long long total, int bs) { return (int)((total + bs - 1) / bs); }

struct Ws {
  float* z; _Float16* xh; _Float16* wt;
  float *el, *er, *m, *s, *exb, *agg;
  float *bnsum, *bnsq, *bnscale, *bnshift;
};

static void run_gat_layer(const float* x, const float* W, const float* al, const float* ar,
                          const float* b, int Fin, int Fout, int H, int D,
                          const int* src, const int* dst, const Ws& w, hipStream_t st) {
  const int N = NNODES, E = NEDGES;
  const int fshift = (Fout == 128) ? 7 : 6;
  const int fmask  = Fout - 1;
  const int dshift = (D == 16) ? 4 : 6;

  cast_f32_f16<<<gblk((long long)N * Fin, 256), 256, 0, st>>>(x, w.xh, N * Fin);
  transpose_cast_W<<<gblk((long long)Fin * Fout, 256), 256, 0, st>>>(W, w.wt, Fin, Fout);

  // N = 100000 = 16 * 6250 exactly; Fin is always 128 -> K fixed at compile time
  const int rowTiles = N / 16;
  gemm_wmma<128><<<gblk(rowTiles, 4), 128, 0, st>>>(w.xh, w.wt, w.z, rowTiles, Fout);

  attn_coeff<<<gblk((long long)N * H, 256), 256, 0, st>>>(w.z, al, ar, w.el, w.er, N, H, D);

  init_ms<<<gblk((long long)N * H, 256), 256, 0, st>>>(w.m, w.s, N * H);
  init_agg_bias<<<gblk((long long)N * Fout, 256), 256, 0, st>>>(w.agg, b, N * Fout, fmask);

  edge_max<<<gblk((long long)E * H, 256), 256, 0, st>>>(src, dst, w.el, w.er, w.m, E, H);
  edge_expsum<<<gblk((long long)E * H, 256), 256, 0, st>>>(src, dst, w.el, w.er, w.m,
                                                           w.exb, w.s, E, H);
  long long total = (long long)E << fshift;
  edge_agg<<<gblk(total, 256), 256, 0, st>>>(src, dst, w.exb, w.s, w.z, w.agg,
                                             total, fshift, fmask, dshift, H);
}

static void run_bn(float* x, const float* g, const float* be, int C, const Ws& w,
                   hipStream_t st) {
  const int N = NNODES;
  zero_f32<<<1, 256, 0, st>>>(w.bnsum, C);
  zero_f32<<<1, 256, 0, st>>>(w.bnsq, C);
  const int blocks = 256, rpb = (N + blocks - 1) / blocks;
  bn_partial<<<blocks, C, 0, st>>>(x, w.bnsum, w.bnsq, N, C, rpb);
  bn_stats<<<1, C, 0, st>>>(w.bnsum, w.bnsq, g, be, w.bnscale, w.bnshift, N, C);
  bn_apply<<<gblk((long long)N * C, 256), 256, 0, st>>>(x, w.bnscale, w.bnshift,
                                                        N * C, C - 1);
}

extern "C" void kernel_launch(void* const* d_in, const int* in_sizes, int n_in,
                              void* d_out, int out_size, void* d_ws, size_t ws_size,
                              hipStream_t stream) {
  (void)in_sizes; (void)n_in; (void)out_size; (void)ws_size;
  const float* h   = (const float*)d_in[0];
  const int*   src = (const int*)d_in[1];
  const int*   dst = (const int*)d_in[2];
  const float *W1 = (const float*)d_in[3],  *al1 = (const float*)d_in[4],
              *ar1 = (const float*)d_in[5], *b1 = (const float*)d_in[6],
              *g1 = (const float*)d_in[7],  *be1 = (const float*)d_in[8];
  const float *W2 = (const float*)d_in[9],  *al2 = (const float*)d_in[10],
              *ar2 = (const float*)d_in[11],*b2 = (const float*)d_in[12],
              *g2 = (const float*)d_in[13], *be2 = (const float*)d_in[14];
  const float *W3 = (const float*)d_in[15], *al3 = (const float*)d_in[16],
              *ar3 = (const float*)d_in[17],*b3 = (const float*)d_in[18];
  float* out = (float*)d_out;

  const size_t N = NNODES, E = NEDGES;
  char* p = (char*)d_ws;
  Ws w;
  w.z   = (float*)p;              p += N * 128 * sizeof(float);      // 51.2 MB
  w.agg = (float*)p;              p += N * 128 * sizeof(float);      // 51.2 MB
  w.exb = (float*)p;              p += E * 8   * sizeof(float);      // 51.2 MB
  w.xh  = (_Float16*)p;           p += N * 128 * sizeof(_Float16);   // 25.6 MB
  w.wt  = (_Float16*)p;           p += 128 * 128 * sizeof(_Float16);
  w.el  = (float*)p;              p += N * 8 * sizeof(float);
  w.er  = (float*)p;              p += N * 8 * sizeof(float);
  w.m   = (float*)p;              p += N * 8 * sizeof(float);
  w.s   = (float*)p;              p += N * 8 * sizeof(float);
  w.bnsum   = (float*)p;          p += 128 * sizeof(float);
  w.bnsq    = (float*)p;          p += 128 * sizeof(float);
  w.bnscale = (float*)p;          p += 128 * sizeof(float);
  w.bnshift = (float*)p;          p += 128 * sizeof(float);

  // Layer 1: [N,128] -> [N,128], 8 heads x 16
  run_gat_layer(h, W1, al1, ar1, b1, 128, 128, 8, 16, src, dst, w, stream);
  run_bn(w.agg, g1, be1, 128, w, stream);

  // Layer 2: [N,128] -> [N,128], 8 heads x 16  (agg is both input and output:
  // it is cast to f16 before being re-initialized inside run_gat_layer)
  run_gat_layer(w.agg, W2, al2, ar2, b2, 128, 128, 8, 16, src, dst, w, stream);
  run_bn(w.agg, g2, be2, 128, w, stream);

  // Layer 3: [N,128] -> [N,64], 1 head x 64, then log_softmax
  run_gat_layer(w.agg, W3, al3, ar3, b3, 128, 64, 1, 64, src, dst, w, stream);
  log_softmax64<<<gblk(NNODES, 4), 128, 0, stream>>>(w.agg, out, NNODES);
}
